// KBestAttentionWeights_7928509629142
// MI455X (gfx1250) — matrix operand order repaired and checked by reading, not compiled, not run on hardware
//
#include <hip/hip_runtime.h>
#include <hip/hip_bf16.h>

// ---------------------------------------------------------------------------
// Problem constants (from the reference)
// ---------------------------------------------------------------------------
constexpr int B   = 2;
constexpr int C   = 64;    // u / g / out channels
constexpr int CA  = 4;     // aux channels
constexpr int H   = 128;
constexpr int W   = 128;
constexpr int EMB = 32;
constexpr int CAP = CA * 9;  // 36 = unfolded patch length
constexpr int KB  = 7;       // top-k
constexpr int HW  = H * W;

// ---------------------------------------------------------------------------
// CDNA5 WMMA plumbing (wave32, v_wmma_f32_16x16x32_bf16)
// ---------------------------------------------------------------------------
typedef __attribute__((ext_vector_type(16))) __bf16    v16bf;
typedef __attribute__((ext_vector_type(8)))  __bf16    v8bf;
typedef __attribute__((ext_vector_type(8)))  float     v8f;
typedef __attribute__((ext_vector_type(4)))  unsigned  v4u;
typedef __attribute__((ext_vector_type(8)))  unsigned  v8u;

__device__ __forceinline__ __bf16 tobf(float f) {
  // round-to-nearest-even fp32 -> bf16
  unsigned u = __builtin_bit_cast(unsigned, f);
  unsigned r = u + 0x7FFFu + ((u >> 16) & 1u);
  unsigned short hbits = (unsigned short)(r >> 16);
  return __builtin_bit_cast(__bf16, hbits);
}

__device__ __forceinline__ v8f wmma_bf16(v16bf a, v16bf b, v8f c) {
  // 8 args: (neg_a, A, neg_b, B, c_mod, C, reuse_a, reuse_b)
  return __builtin_amdgcn_wmma_f32_16x16x32_bf16(false, a, false, b,
                                                 (short)0, c, false, false);
}

__device__ __forceinline__ v16bf combine(v8bf lo, v8bf hi) {
  return __builtin_shufflevector(lo, hi, 0, 1, 2, 3, 4, 5, 6, 7,
                                         8, 9, 10, 11, 12, 13, 14, 15);
}

// A-fragment: 16x32 bf16 tile from row-major LDS W[m][ldk].
// Layout (16-bit A 16x32, matching B's K order): lanes 0-15 hold M=lane,
// K=0..15; lanes 16-31 hold M=lane-16, K=16..31.  16 contiguous bf16 in a
// row-major tile -> two ds_load_b128 per fragment.
__device__ __forceinline__ v16bf load_a_frag(const __bf16* Wm, int lane,
                                             int row0, int kOff, int ldk) {
  const int m  = row0 + (lane & 15);
  const int kb = kOff + ((lane >> 4) ? 16 : 0);
  const __bf16* r = Wm + m * ldk + kb;
  return combine(*(const v8bf*)r, *(const v8bf*)(r + 8));
}

// B-fragment from PIXEL-MAJOR staging X[n][ldn] (K contiguous per pixel).
// Contiguous-K layout used here, same as A (16-bit B 32x16): lanes 0-15 (N=lane) hold K=0..15 in order,
// lanes 16-31 hold K=16..31 -> 16 contiguous bf16 = two ds_load_b128.
// ldn must be a multiple of 8 (16B row alignment).
__device__ __forceinline__ v16bf load_b_fragT(const __bf16* X, int lane,
                                              int kOff, int ldn) {
  const int n  = lane & 15;
  const int kb = kOff + ((lane >> 4) ? 16 : 0);
  const __bf16* p = X + n * ldn + kb;
  return combine(*(const v8bf*)p, *(const v8bf*)(p + 8));
}

// LDS byte-offset of a __shared__ object (generic -> AS(3) -> 32-bit offset)
__device__ __forceinline__ unsigned lds_off_of(const void* p) {
  return (unsigned)(unsigned long long)
      (__attribute__((address_space(3))) const void*)p;
}

// ---------------------------------------------------------------------------
// Kernel 1: patch-embed (VALU, K=36) + phi/theta GEMMs (WMMA, K=32).
// One block = one (b,h) row = 128 pixels; 8 waves x 16-pixel N-tiles.
// emb staged pixel-major (stride 40 halfs = 80B: 16B-aligned, bank-spread).
// ---------------------------------------------------------------------------
__global__ __launch_bounds__(256) void k_embed_phi_theta(
    const float* __restrict__ aux, const float* __restrict__ w_emb,
    const float* __restrict__ w_phi, const float* __restrict__ w_theta,
    float* __restrict__ phi, float* __restrict__ theta) {
  __shared__ float  s_wemb[EMB * CAP];                 // 32x36 fp32
  __shared__ __align__(16) __bf16 s_wphi[EMB * EMB];   // 32x32 bf16 row-major
  __shared__ __align__(16) __bf16 s_wth [EMB * EMB];
  __shared__ __align__(16) __bf16 s_embT[W * 40];      // [pixel][K(32)+pad]

  const int h = blockIdx.x, b = blockIdx.y;
  const int t = threadIdx.x;

  for (int i = t; i < EMB * CAP; i += 256) s_wemb[i] = w_emb[i];
  for (int i = t; i < EMB * EMB; i += 256) {
    s_wphi[i] = tobf(w_phi[i]);
    s_wth [i] = tobf(w_theta[i]);
  }
  __syncthreads();

  const int wv = t >> 5, lane = t & 31;
  const int n = lane & 15, hl = lane >> 4;
  const int x = wv * 16 + n;                  // pixel column (two lanes/pixel)

  // Gather 3x3xCA patch (zero padded), channel index = ca*9 + dy*3 + dx
  float patch[CAP];
#pragma unroll
  for (int ca = 0; ca < CA; ++ca)
#pragma unroll
    for (int dy = 0; dy < 3; ++dy)
#pragma unroll
      for (int dx = 0; dx < 3; ++dx) {
        const int hh = h + dy - 1, ww = x + dx - 1;
        float v = 0.f;
        if (hh >= 0 && hh < H && ww >= 0 && ww < W)
          v = aux[((size_t)(b * CA + ca) * H + hh) * W + ww];
        patch[ca * 9 + dy * 3 + dx] = v;
      }

  // emb: this lane computes 16 of the 32 channels for its pixel, then stores
  // them as two packed 16B ds_store_b128s in pixel-major order.
  v8bf lo, hi;
#pragma unroll
  for (int cc = 0; cc < 16; ++cc) {
    const int c = hl * 16 + cc;
    const float* wr = &s_wemb[c * CAP];
    float acc = 0.f;
#pragma unroll
    for (int i = 0; i < CAP; ++i) acc = fmaf(wr[i], patch[i], acc);
    if (cc < 8) lo[cc] = tobf(acc); else hi[cc - 8] = tobf(acc);
  }
  __bf16* dst = &s_embT[x * 40 + hl * 16];
  *(v8bf*)dst = lo;
  *(v8bf*)(dst + 8) = hi;
  __syncthreads();

  // phi/theta = (32x32) x (32x16) : 2 M-tiles each, single K=32 chunk
  const __bf16* Xw = &s_embT[(wv * 16) * 40];
  const v16bf bf  = load_b_fragT(Xw, lane, 0, 40);
  const v16bf ap0 = load_a_frag(s_wphi, lane, 0,  0, EMB);
  const v16bf ap1 = load_a_frag(s_wphi, lane, 16, 0, EMB);
  const v16bf at0 = load_a_frag(s_wth,  lane, 0,  0, EMB);
  const v16bf at1 = load_a_frag(s_wth,  lane, 16, 0, EMB);
  v8f p0 = {}, p1 = {}, q0 = {}, q1 = {};
  p0 = wmma_bf16(ap0, bf, p0);
  p1 = wmma_bf16(ap1, bf, p1);
  q0 = wmma_bf16(at0, bf, q0);
  q1 = wmma_bf16(at1, bf, q1);

  // D layout: VGPR r, lanes 0-15 -> M=r, lanes 16-31 -> M=r+8 (N = lane&15)
  float* pdst = phi   + (((size_t)b * H + h) * W + x) * EMB;
  float* tdst = theta + (((size_t)b * H + h) * W + x) * EMB;
#pragma unroll
  for (int r = 0; r < 8; ++r) {
    const int m = r + hl * 8;
    pdst[m]      = p0[r];
    pdst[m + 16] = p1[r];
    tdst[m]      = q0[r];
    tdst[m + 16] = q1[r];
  }
}

// ---------------------------------------------------------------------------
// Kernel 2: g = Wg(64x64) x u — dominant GEMM, fully WMMA.
// The u row-tile (64 ch x 128 px fp32 = 32KB) is fetched by the Tensor Data
// Mover as a 2D tile (tile 128x64, dim0_stride = H*W), issued by wave 0,
// tracked with TENSORcnt.  A block-wide pass converts it to the bf16
// pixel-major staging layout (stride 72 halfs = 144B: 16B-aligned).
// 8 waves x 16-px N-tiles; 4 M-tiles x 2 K-chunks = 8 WMMAs per wave.
// ---------------------------------------------------------------------------
__global__ __launch_bounds__(256) void k_g_gemm(
    const float* __restrict__ u, const float* __restrict__ w_g,
    float* __restrict__ g) {
  __shared__ __align__(16) float  s_u32[C * W];   // TDM dest: [c][w] fp32 32KB
  __shared__ __align__(16) __bf16 s_uT [W * 72];  // [pixel][K(64)+pad] bf16
  __shared__ __align__(16) __bf16 s_wg [C * C];   // 64x64 bf16 row-major

  const int h = blockIdx.x, b = blockIdx.y;
  const int t = threadIdx.x;
  const size_t ubase = ((size_t)b * C) * HW + (size_t)h * W;

  // ---- TDM: one DMA per block (wave 0 issues; EXEC is ignored by TDM) ----
  if (t < 32) {
    const unsigned long long ga = (unsigned long long)(u + ubase);
    v4u g0;
    g0[0] = 1u;                                   // count=1, user descriptor
    g0[1] = lds_off_of(s_u32);                    // lds_addr (bytes)
    g0[2] = (unsigned)ga;                         // global_addr[31:0]
    g0[3] = (unsigned)(ga >> 32) | (2u << 30);    // global_addr[56:32]|type=2
    v8u g1;
    g1[0] = 2u << 16;                             // data_size=2 (4 bytes)
    g1[1] = (unsigned)W << 16;                    // tensor_dim0[15:0] = 128
    g1[2] = (unsigned)C << 16;                    // tensor_dim1[15:0] = 64
    g1[3] = (unsigned)W << 16;                    // tile_dim0 = 128
    g1[4] = (unsigned)C;                          // tile_dim1 = 64, tile_dim2=0
    g1[5] = (unsigned)HW;                         // tensor_dim0_stride = 16384
    g1[6] = 0u;                                   // stride[47:32], dim1_stride
    g1[7] = 0u;
    asm volatile("tensor_load_to_lds %0, %1" :: "s"(g0), "s"(g1) : "memory");
  }

  // Weights: global fp32 -> bf16 LDS (overlaps with the TDM transfer)
  for (int i = t; i < C * C; i += 256) s_wg[i] = tobf(w_g[i]);

  if (t < 32) __builtin_amdgcn_s_wait_tensorcnt(0);
  __syncthreads();

  // Convert + transpose: s_u32[c][w] -> s_uT[w][c] (packed 8B ds_store_b64)
  for (int j = t; j < (C / 4) * W; j += 256) {
    const int x  = j & (W - 1);
    const int cg = j >> 7;  // group of 4 channels
    unsigned short pk[4];
#pragma unroll
    for (int e = 0; e < 4; ++e)
      pk[e] = __builtin_bit_cast(unsigned short,
                                 tobf(s_u32[(cg * 4 + e) * W + x]));
    uint2 d;
    d.x = (unsigned)pk[0] | ((unsigned)pk[1] << 16);
    d.y = (unsigned)pk[2] | ((unsigned)pk[3] << 16);
    *(uint2*)&s_uT[x * 72 + cg * 4] = d;
  }
  __syncthreads();

  const int wv = t >> 5, lane = t & 31;
  const int n0 = wv * 16;
  const int n = lane & 15, hl = lane >> 4;

  const __bf16* Xw = &s_uT[n0 * 72];
  const v16bf b0 = load_b_fragT(Xw, lane, 0,  72);
  const v16bf b1 = load_b_fragT(Xw, lane, 32, 72);
#pragma unroll
  for (int mt = 0; mt < 4; ++mt) {
    const v16bf a0 = load_a_frag(s_wg, lane, mt * 16, 0,  C);
    const v16bf a1 = load_a_frag(s_wg, lane, mt * 16, 32, C);
    v8f acc = {};
    acc = wmma_bf16(a0, b0, acc);
    acc = wmma_bf16(a1, b1, acc);
#pragma unroll
    for (int r = 0; r < 8; ++r) {
      const int m = mt * 16 + r + hl * 8;
      g[(((size_t)b * C + m) * H + h) * W + n0 + n] = acc[r];
    }
  }
}

// ---------------------------------------------------------------------------
// Kernel 3: windowed attention (49 taps x dot-32) + top-7 + softmax.
// One thread per pixel; att row lives in LDS so the arg-top-k stays indexable.
// wxy/idx stored as k-major planes [b][k][h][w] for coalesced reads later.
// ---------------------------------------------------------------------------
__global__ __launch_bounds__(128) void k_attn(
    const float* __restrict__ phi, const float* __restrict__ theta,
    float* __restrict__ wxy, int* __restrict__ kidx) {
  __shared__ float s_att[128 * 52];   // stride 52 padding

  const int h = blockIdx.x, b = blockIdx.y;
  const int x = threadIdx.x;

  float4 ph[8];
  const float4* pv =
      (const float4*)(phi + (((size_t)b * H + h) * W + x) * EMB);
#pragma unroll
  for (int i = 0; i < 8; ++i) ph[i] = pv[i];

  float* att = &s_att[x * 52];
  for (int dy = 0; dy < 7; ++dy) {
    const int hh = h + dy - 3;
    for (int dx = 0; dx < 7; ++dx) {
      const int ww = x + dx - 3;
      float a = 0.f;
      if (hh >= 0 && hh < H && ww >= 0 && ww < W) {
        const float4* tv =
            (const float4*)(theta + (((size_t)b * H + hh) * W + ww) * EMB);
#pragma unroll
        for (int i = 0; i < 8; ++i) {
          const float4 tw = tv[i];
          a = fmaf(ph[i].x, tw.x, a);
          a = fmaf(ph[i].y, tw.y, a);
          a = fmaf(ph[i].z, tw.z, a);
          a = fmaf(ph[i].w, tw.w, a);
        }
      }
      att[dy * 7 + dx] = a;   // zero-padded taps contribute att = 0 (matches ref)
    }
  }

  // top-7 by repeated argmax (ties -> lowest index, like lax.top_k)
  float bv[KB]; int bi[KB];
#pragma unroll
  for (int s = 0; s < KB; ++s) {
    float best = -3.4e38f; int bidx = 0;
    for (int k = 0; k < 49; ++k) {
      const float v = att[k];
      if (v > best) { best = v; bidx = k; }
    }
    bv[s] = best; bi[s] = bidx;
    att[bidx] = -3.4e38f;
  }

  // softmax over the 7 kept taps (bv[0] is the max)
  float e[KB], sum = 0.f;
#pragma unroll
  for (int s = 0; s < KB; ++s) { e[s] = __expf(bv[s] - bv[0]); sum += e[s]; }
  const float inv = 1.f / sum;
#pragma unroll
  for (int s = 0; s < KB; ++s) {
    const size_t p = (((size_t)b * KB + s) * H + h) * W + x;
    wxy[p]  = e[s] * inv;
    kidx[p] = bi[s];
  }
}

// ---------------------------------------------------------------------------
// Kernel 4: out[b,c,h,w] = sum_k wxy[k] * g[b,c,h+dy_k,w+dx_k]  (bchw output)
// ---------------------------------------------------------------------------
__global__ __launch_bounds__(128) void k_gather(
    const float* __restrict__ g, const float* __restrict__ wxy,
    const int* __restrict__ kidx, float* __restrict__ out) {
  const int h = blockIdx.x, c = blockIdx.y, b = blockIdx.z;
  const int x = threadIdx.x;

  // Warm the k-major weight/index planes (global_prefetch_b8)
  __builtin_prefetch(&wxy[(((size_t)b * KB) * H + h) * W + x], 0, 1);
  __builtin_prefetch(&kidx[(((size_t)b * KB) * H + h) * W + x], 0, 1);

  float acc = 0.f;
#pragma unroll
  for (int s = 0; s < KB; ++s) {
    const size_t p = (((size_t)b * KB + s) * H + h) * W + x;
    const int   kk = kidx[p];
    const float wg = wxy[p];
    const int dy = kk / 7 - 3, dx = kk % 7 - 3;
    const int hh = h + dy, ww = x + dx;
    float v = 0.f;
    if (hh >= 0 && hh < H && ww >= 0 && ww < W)
      v = g[(((size_t)b * C + c) * H + hh) * W + ww];
    acc = fmaf(wg, v, acc);
  }
  out[(((size_t)b * C + c) * H + h) * W + x] = acc;
}

// ---------------------------------------------------------------------------
// Launch: all on `stream`, graph-capture safe (no alloc/sync/memcpy).
// Workspace map (bytes):
//   [0,        4 MB) phi   f32 [b][h][w][32]
//   [4 MB,     8 MB) theta f32 [b][h][w][32]
//   [8 MB,    16 MB) g     f32 bchw
//   [16 MB, +896 KB) wxy   f32 [b][7][h][w]
//   [.. , +896 KB)   idx   i32 [b][7][h][w]
// ---------------------------------------------------------------------------
extern "C" void kernel_launch(void* const* d_in, const int* in_sizes, int n_in,
                              void* d_out, int out_size, void* d_ws,
                              size_t ws_size, hipStream_t stream) {
  (void)in_sizes; (void)n_in; (void)out_size; (void)ws_size;

  const float* u       = (const float*)d_in[0];
  const float* aux     = (const float*)d_in[1];
  const float* w_emb   = (const float*)d_in[2];
  const float* w_g     = (const float*)d_in[3];
  const float* w_phi   = (const float*)d_in[4];
  const float* w_theta = (const float*)d_in[5];
  float* out = (float*)d_out;

  char* ws = (char*)d_ws;
  constexpr size_t SZ_PT  = (size_t)B * HW * EMB * sizeof(float);  // 4 MB
  constexpr size_t SZ_G   = (size_t)B * C  * HW * sizeof(float);   // 8 MB
  constexpr size_t SZ_K   = (size_t)B * KB * HW * sizeof(float);   // 896 KB
  float* phi   = (float*)(ws);
  float* theta = (float*)(ws + SZ_PT);
  float* g     = (float*)(ws + 2 * SZ_PT);
  float* wxy   = (float*)(ws + 2 * SZ_PT + SZ_G);
  int*   kidx  = (int*)  (ws + 2 * SZ_PT + SZ_G + SZ_K);

  const dim3 rows(H, B);
  k_embed_phi_theta<<<rows, 256, 0, stream>>>(aux, w_emb, w_phi, w_theta,
                                              phi, theta);
  k_g_gemm<<<rows, 256, 0, stream>>>(u, w_g, g);
  k_attn<<<rows, 128, 0, stream>>>(phi, theta, wxy, kidx);
  k_gather<<<dim3(H, C, B), 128, 0, stream>>>(g, wxy, kidx, out);
}
